// GIN_1812476199284
// MI455X (gfx1250) — compile-verified, matrix operand order checked by reading, not run on hardware
//
#include <hip/hip_runtime.h>

#define NN     50000      // nodes
#define NE     800000     // edges
#define INC    165
#define HIDC   256
#define OUTC   2
#define K1PAD  192        // 165 padded to multiple of 32
#define MPAD   50016      // 50000 padded to multiple of 16

typedef __attribute__((ext_vector_type(16))) __bf16 v16bf;
typedef __attribute__((ext_vector_type(8)))  __bf16 v8bf;
typedef __attribute__((ext_vector_type(8)))  float  v8f;

__device__ __forceinline__ unsigned short f2bf(float f) {
  unsigned int u = __builtin_bit_cast(unsigned int, f);
  u += 0x7FFFu + ((u >> 16) & 1u);          // round-to-nearest-even
  return (unsigned short)(u >> 16);
}

// ---------------------------------------------------------------------------
// Pack weight W[K x 256] (f32, row-major) into per-lane WMMA bf16 B-fragments:
// P[ntile][kstep][lane][16] ; col = ntile*16 + lane%16, kb = kstep*32 + (lane/16)*8
// halves 0..7 -> K = kb..kb+7 ; halves 8..15 -> K = kb+16..kb+23. Zero-pad K>=Kreal.
// ---------------------------------------------------------------------------
__global__ void pack_b_kernel(const float* __restrict__ W, int Kreal, int ksteps,
                              unsigned short* __restrict__ P) {
  int t    = blockIdx.x;            // ntile * ksteps + kstep
  int lane = threadIdx.x;           // 0..31
  int ntile = t / ksteps;
  int kstep = t % ksteps;
  int col = ntile * 16 + (lane & 15);
  int kb  = kstep * 32 + ((lane >> 4) << 3);
  unsigned short* p = P + (((size_t)t) * 32 + lane) * 16;
#pragma unroll
  for (int j = 0; j < 8; ++j) {
    int k1 = kb + j, k2 = kb + 16 + j;
    p[j]     = (k1 < Kreal) ? f2bf(W[(size_t)k1 * HIDC + col]) : (unsigned short)0;
    p[8 + j] = (k2 < Kreal) ? f2bf(W[(size_t)k2 * HIDC + col]) : (unsigned short)0;
  }
}

// agg1[i][k] = x[i][k] (self term of GIN) over MPAD x 192, zero padding.
__global__ void init_agg1_kernel(const float* __restrict__ x, float* __restrict__ agg) {
  size_t tid = (size_t)blockIdx.x * blockDim.x + threadIdx.x;
  if (tid >= (size_t)MPAD * K1PAD) return;
  int i = (int)(tid / K1PAD), k = (int)(tid % K1PAD);
  agg[tid] = (i < NN && k < INC) ? x[(size_t)i * INC + k] : 0.0f;
}

// Edge scatter: agg1[dst] += x[src], 165 channels, f32 atomics (exact-order-free sum).
__global__ void scatter1_kernel(const float* __restrict__ x, const int* __restrict__ ei,
                                float* __restrict__ agg) {
  int e = blockIdx.x;
  int s = ei[e];            // edge_index[0][e]
  int d = ei[NE + e];       // edge_index[1][e]
  int k = threadIdx.x;
  if (k < INC) atomicAdd(&agg[(size_t)d * K1PAD + k], x[(size_t)s * INC + k]);
}

__global__ void scatter2_kernel(const float* __restrict__ h, const int* __restrict__ ei,
                                float* __restrict__ agg) {
  int e = blockIdx.x;
  int s = ei[e];
  int d = ei[NE + e];
  int k = threadIdx.x;
  atomicAdd(&agg[(size_t)d * HIDC + k], h[(size_t)s * HIDC + k]);
}

__global__ void cvt_bf16_kernel(const float* __restrict__ in,
                                unsigned short* __restrict__ out, size_t n) {
  size_t i = (size_t)blockIdx.x * blockDim.x + threadIdx.x;
  if (i < n) out[i] = f2bf(in[i]);
}

// ---------------------------------------------------------------------------
// WMMA GEMM: C[MPAD x 256] = relu?(A[MPAD x Kpad]_bf16 @ Bpacked + bias)
// One wave = one 16x16 tile. Block = 256 thr = 8 waves -> 8 N-tiles; grid.y=2 covers N=256.
// ---------------------------------------------------------------------------
__global__ __launch_bounds__(256) void gin_gemm_wmma(
    const unsigned short* __restrict__ A,
    const unsigned short* __restrict__ Bp,
    const float* __restrict__ bias,
    int Kpad,
    unsigned short* __restrict__ out_bf,   // nullable
    float* __restrict__ out_f,             // nullable
    float* __restrict__ out_f2,            // nullable (dup write)
    int relu)
{
  const int lane   = threadIdx.x & 31;
  const int wave   = threadIdx.x >> 5;
  const int mtile  = blockIdx.x;
  const int ntile  = blockIdx.y * 8 + wave;
  const int ksteps = Kpad >> 5;

  const int row = mtile * 16 + (lane & 15);
  const int col = ntile * 16 + (lane & 15);
  // A fragment: halves 0..7 at K = kb..kb+7, halves 8..15 at K = kb+16..kb+23,
  // kb = k0 + (lane/16)*8  (ISA 16-bit A-matrix 16x32 layout)
  const unsigned short* arow = A + (size_t)row * Kpad + ((lane >> 4) << 3);
  const unsigned short* bptr = Bp + (((size_t)ntile * ksteps) * 32 + lane) * 16;

  float bv = bias[col];
  v8f acc;
#pragma unroll
  for (int i = 0; i < 8; ++i) acc[i] = bv;

  for (int ks = 0; ks < ksteps; ++ks) {
    v8bf alo = *(const v8bf*)(arow);
    v8bf ahi = *(const v8bf*)(arow + 16);
    v16bf av = __builtin_shufflevector(alo, ahi,
        0,1,2,3,4,5,6,7,8,9,10,11,12,13,14,15);
    v16bf bfrag = *(const v16bf*)(bptr);
    acc = __builtin_amdgcn_wmma_f32_16x16x32_bf16(
        /*neg_a=*/false, av, /*neg_b=*/false, bfrag,
        /*c_mod=*/(short)0, acc, /*reuse_a=*/false, /*reuse_b=*/false);
    arow += 32;
    bptr += 32 * 16;
  }

  // C/D layout: VGPR i -> M = i (lanes 0-15) or i+8 (lanes 16-31), N = lane%16
  const int mbase = mtile * 16 + ((lane >> 4) << 3);
#pragma unroll
  for (int i = 0; i < 8; ++i) {
    float v = acc[i];
    if (relu) v = v > 0.0f ? v : 0.0f;
    size_t off = (size_t)(mbase + i) * HIDC + col;
    if (out_bf) out_bf[off] = f2bf(v);
    if (out_f)  out_f[off]  = v;
    if (out_f2) out_f2[off] = v;
  }
}

// Final 256 -> 2 projection (too skinny for WMMA): one thread per node.
__global__ void final_proj_kernel(const float* __restrict__ h,
                                  const float* __restrict__ W,
                                  const float* __restrict__ b,
                                  float* __restrict__ out) {
  int i = blockIdx.x * blockDim.x + threadIdx.x;
  if (i >= NN) return;
  float a0 = b[0], a1 = b[1];
  const float* row = h + (size_t)i * HIDC;
#pragma unroll 8
  for (int k = 0; k < HIDC; ++k) {
    float v = row[k];
    a0 += v * W[2 * k];
    a1 += v * W[2 * k + 1];
  }
  out[2 * i]     = a0;
  out[2 * i + 1] = a1;
}

extern "C" void kernel_launch(void* const* d_in, const int* in_sizes, int n_in,
                              void* d_out, int out_size, void* d_ws, size_t ws_size,
                              hipStream_t stream) {
  const float* x   = (const float*)d_in[0];
  const int*   ei  = (const int*)d_in[1];
  const float* W1a = (const float*)d_in[2];
  const float* b1a = (const float*)d_in[3];
  const float* W1b = (const float*)d_in[4];
  const float* b1b = (const float*)d_in[5];
  const float* W2a = (const float*)d_in[6];
  const float* b2a = (const float*)d_in[7];
  const float* W2b = (const float*)d_in[8];
  const float* b2b = (const float*)d_in[9];
  float* out = (float*)d_out;

  char* base = (char*)d_ws;
  size_t off = 0;
  auto take = [&](size_t bytes) -> char* {
    off = (off + 255) & ~(size_t)255;
    char* p = base + off;
    off += bytes;
    return p;
  };

  unsigned short* w1a_p = (unsigned short*)take((size_t)16 * 6 * 32 * 16 * 2);
  unsigned short* w1b_p = (unsigned short*)take((size_t)16 * 8 * 32 * 16 * 2);
  unsigned short* w2a_p = (unsigned short*)take((size_t)16 * 8 * 32 * 16 * 2);
  float*          agg1  = (float*)take((size_t)MPAD * K1PAD * 4);
  unsigned short* z1bf  = (unsigned short*)take((size_t)MPAD * K1PAD * 2);
  unsigned short* h1r   = (unsigned short*)take((size_t)MPAD * HIDC * 2);
  float*          h1    = (float*)take((size_t)MPAD * HIDC * 4);
  float*          agg2  = (float*)take((size_t)MPAD * HIDC * 4);
  // Aliased reuse (sequential dependencies make this safe):
  float*          h2r   = agg1;   // agg1(38.4M)+z1bf(19.2M) contiguous >= 51.2M needed
  unsigned short* z2bf  = h1r;    // same size, h1r dead after GEMM1b

  // 1) pack weights to WMMA B-fragment order (bf16)
  pack_b_kernel<<<16 * (K1PAD / 32), 32, 0, stream>>>(W1a, INC,  K1PAD / 32, w1a_p);
  pack_b_kernel<<<16 * (HIDC  / 32), 32, 0, stream>>>(W1b, HIDC, HIDC  / 32, w1b_p);
  pack_b_kernel<<<16 * (HIDC  / 32), 32, 0, stream>>>(W2a, HIDC, HIDC  / 32, w2a_p);

  // 2) layer-1 aggregation: z1 = x + scatter_add(x[src] -> dst)
  {
    size_t n = (size_t)MPAD * K1PAD;
    init_agg1_kernel<<<(unsigned)((n + 255) / 256), 256, 0, stream>>>(x, agg1);
  }
  scatter1_kernel<<<NE, 192, 0, stream>>>(x, ei, agg1);
  {
    size_t n = (size_t)MPAD * K1PAD;
    cvt_bf16_kernel<<<(unsigned)((n + 255) / 256), 256, 0, stream>>>(agg1, z1bf, n);
  }

  // 3) MLP1: h1r = relu(z1 @ W1a + b1a) [bf16]; h1 = relu(h1r @ W1b + b1b) [f32]
  dim3 ggrid(MPAD / 16, 2, 1);
  gin_gemm_wmma<<<ggrid, 256, 0, stream>>>(z1bf, w1a_p, b1a, K1PAD,
                                           h1r, nullptr, nullptr, 1);
  //    dual-write: h1 (read by scatter) and agg2 (self term init for layer 2)
  gin_gemm_wmma<<<ggrid, 256, 0, stream>>>(h1r, w1b_p, b1b, HIDC,
                                           nullptr, h1, agg2, 1);

  // 4) layer-2 aggregation: z2 = h1 + scatter_add(h1[src] -> dst)
  scatter2_kernel<<<NE, HIDC, 0, stream>>>(h1, ei, agg2);
  {
    size_t n = (size_t)MPAD * HIDC;
    cvt_bf16_kernel<<<(unsigned)((n + 255) / 256), 256, 0, stream>>>(agg2, z2bf, n);
  }

  // 5) MLP2 first layer: h2r = relu(z2 @ W2a + b2a) [f32]
  gin_gemm_wmma<<<ggrid, 256, 0, stream>>>(z2bf, w2a_p, b2a, HIDC,
                                           nullptr, h2r, nullptr, 1);

  // 6) final 256 -> 2 projection
  final_proj_kernel<<<(NN + 255) / 256, 256, 0, stream>>>(h2r, W2b, b2b, out);
}
